// Self_Atten_57810259804665
// MI455X (gfx1250) — matrix operand order repaired and checked by reading, not compiled, not run on hardware
//
#include <hip/hip_runtime.h>
#include <hip/hip_bf16.h>

typedef __attribute__((ext_vector_type(16))) _Float16 v16h;
typedef __attribute__((ext_vector_type(8)))  _Float16 v8h;
typedef __attribute__((ext_vector_type(8)))  float    v8f;

#define EPI_F16_COL        0   // D[n*ldd + m] f16 (K-major for next GEMM), +bias[m]
#define EPI_F16_ROW        1   // D[m*ldd + n] f16, +bias[m]
#define EPI_EXP_F16_ROW    2   // D[m*ldd + n] f16 = exp(clip(x)); rowsum[m] += sum_n
#define EPI_F32_SCALED_ROW 3   // D[m*ldd + n] f32 = x / colscale[n]

union FragA { v16h v; v8h h[2]; };

__device__ __forceinline__ v8f wmma_f16(v16h a, v16h b, v8f c) {
  return __builtin_amdgcn_wmma_f32_16x16x32_f16(false, a, false, b, (short)0, c,
                                                false, false);
}

// CDNA5 async global->LDS copy (ISA 15.18.3, GLOBAL_LOAD_ASYNC_TO_LDS_B128).
__device__ __forceinline__ void async_copy_b128(unsigned lds_addr, const void* gaddr) {
  asm volatile("global_load_async_to_lds_b128 %0, %1, off"
               : : "v"(lds_addr), "v"(gaddr) : "memory");
}

template<int N>
__device__ __forceinline__ void wait_asynccnt() {
#if __has_builtin(__builtin_amdgcn_s_wait_asynccnt)
  __builtin_amdgcn_s_wait_asynccnt(N);
#else
  asm volatile("s_wait_asynccnt %0" : : "i"(N) : "memory");
#endif
}

// Tiled GEMM: D = A(MxK, row-major) * B(KxN, stored K-major as [N][K]).
// Block tile BM x 128, 8 waves; each wave: AF(=BM/64) A-frags x 4 B-frags.
// Async double-buffered LDS staging, last iteration peeled.
template<int EPI, int BM>
__global__ __launch_bounds__(256) void wmma_gemm(
    const _Float16* __restrict__ A, long long strideA, int lda,
    const _Float16* __restrict__ B, long long strideB, int ldb,
    void* __restrict__ D, long long strideD, int ldd,
    const float* __restrict__ bias, const float* __restrict__ colscale,
    float* __restrict__ rowsum, int K)
{
  constexpr int AF = BM / 64;          // A fragments per wave (1 or 2)
  __shared__ __align__(32) _Float16 As[2][BM * 32];
  __shared__ __align__(32) _Float16 Bs[2][128 * 32];
  __shared__ float red[BM];

  const int m0 = blockIdx.y * BM;
  const int n0 = blockIdx.x * 128;
  const _Float16* Ab = A + (long long)blockIdx.z * strideA;
  const _Float16* Bb = B + (long long)blockIdx.z * strideB;

  const int t    = threadIdx.x;
  const int lane = t & 31;
  const int wid  = t >> 5;
  const int wy   = wid & 3;    // M subtile group
  const int wx   = wid >> 2;   // N 64-half

  const int cc = (t & 3) * 8;  // copy col (halves), 16B chunk per op

  auto issue = [&](int buf, int k0) {
    unsigned la = (unsigned)(size_t)&As[buf][0];
    unsigned lb = (unsigned)(size_t)&Bs[buf][0];
#pragma unroll
    for (int i = 0; i < AF; ++i) {
      const int r = (t + i * 256) >> 2;
      async_copy_b128(la + (unsigned)((r * 32 + cc) * 2),
                      Ab + (size_t)(m0 + r) * lda + k0 + cc);
    }
#pragma unroll
    for (int i = 0; i < 2; ++i) {
      const int r = (t + i * 256) >> 2;
      async_copy_b128(lb + (unsigned)((r * 32 + cc) * 2),
                      Bb + (size_t)(n0 + r) * ldb + k0 + cc);
    }
  };

  v8f acc[AF][4] = {};

  auto compute = [&](int cur) {
    const _Float16* as_ = &As[cur][0];
    const _Float16* bs_ = &Bs[cur][0];
    const int kb   = (lane >> 4) * 16;
    const int krow = (lane >> 4) * 16;
    const int bc   = wx * 64 + (lane & 15);
    v16h bf[4];
#pragma unroll
    for (int j = 0; j < 4; ++j)
      bf[j] = *(const v16h*)(bs_ + (bc + j * 16) * 32 + krow);
#pragma unroll
    for (int i = 0; i < AF; ++i) {
      FragA af;
      const int arow = wy * (16 * AF) + i * 16 + (lane & 15);
      af.h[0] = *(const v8h*)(as_ + arow * 32 + kb);
      af.h[1] = *(const v8h*)(as_ + arow * 32 + 8 + kb);
#pragma unroll
      for (int j = 0; j < 4; ++j)
        acc[i][j] = wmma_f16(af.v, bf[j], acc[i][j]);
    }
  };

  const int nk = K >> 5;
  issue(0, 0);
  for (int kt = 0; kt < nk - 1; ++kt) {
    const int cur = kt & 1;
    issue(cur ^ 1, (kt + 1) * 32);
    wait_asynccnt<AF + 2>();
    __syncthreads();
    compute(cur);
    __syncthreads();
  }
  wait_asynccnt<0>();
  __syncthreads();
  compute((nk - 1) & 1);

  // D fragment: lane -> local row 8*(lane>>4)+r (per 16-row frag), col lane&15
  const int nc0    = n0 + wx * 64 + (lane & 15);
  const int mbase0 = m0 + wy * (16 * AF) + ((lane >> 4) << 3);

  float sc[4];
  if (EPI == EPI_F32_SCALED_ROW) {
    const float* cs = colscale + (size_t)blockIdx.z * 1024;
#pragma unroll
    for (int j = 0; j < 4; ++j) sc[j] = 1.0f / cs[nc0 + j * 16];
  }
  if (EPI == EPI_EXP_F16_ROW) {
    if (t < BM) red[t] = 0.0f;
    __syncthreads();
  }

  float rpart[AF][8];
#pragma unroll
  for (int i = 0; i < AF; ++i) {
    const int mb = mbase0 + i * 16;
    if (EPI == EPI_F16_COL) {
      _Float16* d = (_Float16*)D + (long long)blockIdx.z * strideD;
      float bb[8];
#pragma unroll
      for (int r = 0; r < 8; ++r) bb[r] = bias ? bias[mb + r] : 0.0f;
#pragma unroll
      for (int j = 0; j < 4; ++j) {
        v8h hv;
#pragma unroll
        for (int r = 0; r < 8; ++r) hv[r] = (_Float16)(acc[i][j][r] + bb[r]);
        *(v8h*)(d + (size_t)(nc0 + j * 16) * ldd + mb) = hv;
      }
    } else if (EPI == EPI_F16_ROW) {
      _Float16* d = (_Float16*)D + (long long)blockIdx.z * strideD;
#pragma unroll
      for (int r = 0; r < 8; ++r) {
        const int m = mb + r;
        const float bbv = bias ? bias[m] : 0.0f;
#pragma unroll
        for (int j = 0; j < 4; ++j)
          d[(size_t)m * ldd + nc0 + j * 16] = (_Float16)(acc[i][j][r] + bbv);
      }
    } else if (EPI == EPI_EXP_F16_ROW) {
      _Float16* d = (_Float16*)D + (long long)blockIdx.z * strideD;
#pragma unroll
      for (int r = 0; r < 8; ++r) {
        const int m = mb + r;
        float s = 0.0f;
#pragma unroll
        for (int j = 0; j < 4; ++j) {
          float xv = __expf(fminf(1.0f, fmaxf(-1.0f, acc[i][j][r])));
          d[(size_t)m * ldd + nc0 + j * 16] = (_Float16)xv;
          s += xv;
        }
        rpart[i][r] = s;
      }
    } else {
      float* d = (float*)D + (long long)blockIdx.z * strideD;
#pragma unroll
      for (int r = 0; r < 8; ++r) {
        const int m = mb + r;
#pragma unroll
        for (int j = 0; j < 4; ++j)
          d[(size_t)m * ldd + nc0 + j * 16] = acc[i][j][r] * sc[j];
      }
    }
  }

  if (EPI == EPI_EXP_F16_ROW) {
    const int mloc = mbase0 - m0;
#pragma unroll
    for (int i = 0; i < AF; ++i)
#pragma unroll
      for (int r = 0; r < 8; ++r)
        atomicAdd(&red[mloc + i * 16 + r], rpart[i][r]);
    __syncthreads();
    if (t < BM)
      atomicAdd(rowsum + (size_t)blockIdx.z * 1024 + m0 + t, red[t]);
  }
}

// im2col: Pt[b][n][k] f16, k = ci*256 + ky*16 + kx, n = py*32 + px
__global__ __launch_bounds__(256) void im2col_f16(const float* __restrict__ x,
                                                  _Float16* __restrict__ Pt,
                                                  int total) {
  int idx = blockIdx.x * 256 + threadIdx.x;
  if (idx >= total) return;
  int k = idx % 768;
  int n = (idx / 768) & 1023;
  int b = idx / (768 * 1024);
  int ci = k >> 8, rr = k & 255, ky = rr >> 4, kx = rr & 15;
  int py = n >> 5, px = n & 31;
  float v = x[(((size_t)b * 3 + ci) * 512 + (py * 16 + ky)) * 512 + (px * 16 + kx)];
  Pt[idx] = (_Float16)v;
}

__global__ __launch_bounds__(256) void f32_to_f16(const float* __restrict__ s,
                                                  _Float16* __restrict__ d, int n) {
  int i = blockIdx.x * 256 + threadIdx.x;
  if (i < n) d[i] = (_Float16)s[i];
}

__global__ __launch_bounds__(256) void zero_f32(float* __restrict__ p, int n) {
  int i = blockIdx.x * 256 + threadIdx.x;
  if (i < n) p[i] = 0.0f;
}

extern "C" void kernel_launch(void* const* d_in, const int* in_sizes, int n_in,
                              void* d_out, int out_size, void* d_ws, size_t ws_size,
                              hipStream_t stream) {
  (void)in_sizes; (void)n_in; (void)out_size; (void)ws_size;
  const float* x   = (const float*)d_in[0];
  const float* wc  = (const float*)d_in[1];
  const float* wqf = (const float*)d_in[2];
  const float* bq  = (const float*)d_in[3];
  const float* wkf = (const float*)d_in[4];
  const float* bk  = (const float*)d_in[5];
  const float* wvf = (const float*)d_in[6];
  const float* bv  = (const float*)d_in[7];
  float* out = (float*)d_out;

  char* w = (char*)d_ws;
  _Float16* wc16 = (_Float16*)w; w += (size_t)512 * 768 * 2;
  _Float16* wq16 = (_Float16*)w; w += (size_t)64 * 512 * 2;
  _Float16* wk16 = (_Float16*)w; w += (size_t)64 * 512 * 2;
  _Float16* wv16 = (_Float16*)w; w += (size_t)512 * 512 * 2;
  _Float16* Pt   = (_Float16*)w; w += (size_t)32 * 1024 * 768 * 2;   // [b][n][768]
  _Float16* Xt   = (_Float16*)w; w += (size_t)32 * 1024 * 512 * 2;   // [b][n][512]
  _Float16* Qt   = (_Float16*)w; w += (size_t)32 * 1024 * 64 * 2;    // [b][n][64]
  _Float16* Kt   = (_Float16*)w; w += (size_t)32 * 1024 * 64 * 2;    // [b][m][64]
  _Float16* Vr   = (_Float16*)w; w += (size_t)32 * 512 * 1024 * 2;   // [b][c][m]
  _Float16* At   = (_Float16*)w; w += (size_t)32 * 1024 * 1024 * 2;  // [b][n][m] exp
  float*    rs   = (float*)w;    w += (size_t)32 * 1024 * 4;         // row sums

  f32_to_f16<<<(512 * 768 + 255) / 256, 256, 0, stream>>>(wc, wc16, 512 * 768);
  f32_to_f16<<<(64 * 512 + 255) / 256, 256, 0, stream>>>(wqf, wq16, 64 * 512);
  f32_to_f16<<<(64 * 512 + 255) / 256, 256, 0, stream>>>(wkf, wk16, 64 * 512);
  f32_to_f16<<<(512 * 512 + 255) / 256, 256, 0, stream>>>(wvf, wv16, 512 * 512);
  zero_f32<<<(32 * 1024 + 255) / 256, 256, 0, stream>>>(rs, 32 * 1024);

  {
    int total = 32 * 1024 * 768;
    im2col_f16<<<(total + 255) / 256, 256, 0, stream>>>(x, Pt, total);
  }

  // X = Wc @ P  -> Xt[n][c]
  wmma_gemm<EPI_F16_COL, 128><<<dim3(8, 4, 32), 256, 0, stream>>>(
      wc16, 0, 768, Pt, (long long)1024 * 768, 768,
      Xt, (long long)1024 * 512, 512, nullptr, nullptr, nullptr, 768);
  // Q = wq @ X -> Qt[n][64]
  wmma_gemm<EPI_F16_COL, 64><<<dim3(8, 1, 32), 256, 0, stream>>>(
      wq16, 0, 512, Xt, (long long)1024 * 512, 512,
      Qt, (long long)1024 * 64, 64, bq, nullptr, nullptr, 512);
  // K = wk @ X -> Kt[m][64]
  wmma_gemm<EPI_F16_COL, 64><<<dim3(8, 1, 32), 256, 0, stream>>>(
      wk16, 0, 512, Xt, (long long)1024 * 512, 512,
      Kt, (long long)1024 * 64, 64, bk, nullptr, nullptr, 512);
  // V = wv @ X -> Vr[c][m] row-major
  wmma_gemm<EPI_F16_ROW, 128><<<dim3(8, 4, 32), 256, 0, stream>>>(
      wv16, 0, 512, Xt, (long long)1024 * 512, 512,
      Vr, (long long)512 * 1024, 1024, bv, nullptr, nullptr, 512);
  // attn_unnorm[n][m] = exp(clip(Qt @ Kt^T)); rowsum[n] accumulated
  wmma_gemm<EPI_EXP_F16_ROW, 128><<<dim3(8, 8, 32), 256, 0, stream>>>(
      Qt, (long long)1024 * 64, 64, Kt, (long long)1024 * 64, 64,
      At, (long long)1024 * 1024, 1024, nullptr, nullptr, rs, 64);
  // out[c][n] = (V @ attn_unnorm^T) * (1/rowsum[n])
  wmma_gemm<EPI_F32_SCALED_ROW, 128><<<dim3(8, 4, 32), 256, 0, stream>>>(
      Vr, (long long)512 * 1024, 1024, At, (long long)1024 * 1024, 1024,
      out, (long long)512 * 1024, 1024, nullptr, rs, nullptr, 1024);
}
